// VA_42219528519915
// MI455X (gfx1250) — matrix operand order repaired and checked by reading, not compiled, not run on hardware
//
#include <hip/hip_runtime.h>
#include <hip/hip_bf16.h>

// B,N,L,C,CA = 16,256,64,256,64 ; one workgroup per (b,l) pair.
// bf16 WMMA (16x16x32, f32 accum) for all six GEMMs (Q, K, V, S, O, F).
// This revision software-pipelines LDS B-fragment loads so the backend can
// use partial s_wait_dscnt instead of draining before every v_wmma.

namespace {

constexpr int kB = 16, kN = 256, kL = 64, kC = 256, kA = 64;
constexpr int kThreads = 512; // 16 waves of 32

typedef __attribute__((ext_vector_type(16))) __bf16 v16bf;
typedef __attribute__((ext_vector_type(8)))  __bf16 v8bf;
typedef __attribute__((ext_vector_type(8)))  float  v8f;

// LDS layout (bytes), total 208,896:
//   [0,      101376)  phase1: Wq_t / Wk_t / Wv_t  (each 64 x 264 bf16 = 33,792)
//                     phase>=3: stage  = [0, 36864)   16 waves x 2304B (16x72 bf16)
//                               fcs    = [36864, 73728) 256 x 72 bf16
//   [101376, 138240)  Qs  256 x 72 bf16 (row-major)
//   [138240, 175104)  Ks  256 x 72 bf16 (row-major)
//   [175104, 208896)  Vt   64 x 264 bf16 (transposed: [a][n])
constexpr int WT_STRIDE = 264;  // bf16 elems per row of W_t / Vt
constexpr int QK_STRIDE = 72;   // bf16 elems per row of Qs / Ks / fcs / O-stage
constexpr int STG_P_STRIDE = 40;
constexpr int LDS_BYTES = 208896;

__device__ inline v16bf cat16(v8bf lo, v8bf hi) {
  return __builtin_shufflevector(lo, hi, 0,1,2,3,4,5,6,7,8,9,10,11,12,13,14,15);
}

// A-fragment (16x32 bf16) from a row-major LDS row.
// Lane m (0..15)/(16..31 -> g=1): elems 0..7 = row[kb+g*8..+7],
// elems 8..15 = row[kb+16+g*8..+7] (matches documented A-table).
__device__ inline v16bf load_a(const __bf16* row, int kb, int g) {
  v8bf lo = *(const v8bf*)(row + kb + g * 8);
  v8bf hi = *(const v8bf*)(row + kb + 16 + g * 8);
  return cat16(lo, hi);
}

// B-fragment (32x16 bf16): lane (g,i) holds col i, elems 0..7 = K kb+g*8..+7,
// elems 8..15 = K kb+16+g*8..+7 (off = kb+g*16; matches A-table order).
__device__ inline v16bf load_b(const __bf16* row, int off) {
  v8bf lo = *(const v8bf*)(row + off - ((off & 16) >> 1));
  v8bf hi = *(const v8bf*)(row + off - ((off & 16) >> 1) + 16);
  return cat16(lo, hi);
}

__device__ inline v8f wmma_bf16(v16bf a, v16bf b, v8f c) {
  return __builtin_amdgcn_wmma_f32_16x16x32_bf16(
      /*neg_a=*/false, a, /*neg_b=*/false, b,
      /*c_mod=*/(short)0, c, /*reuse_a=*/false, /*reuse_b=*/false);
}

} // namespace

__global__ __launch_bounds__(kThreads, 1)
void va_attn_kernel(const float* __restrict__ Y,
                    const float* __restrict__ Wq_g,
                    const float* __restrict__ Wk_g,
                    const float* __restrict__ Wv_g,
                    const float* __restrict__ Qb_g,
                    const float* __restrict__ Kb_g,
                    const float* __restrict__ Vb_g,
                    const float* __restrict__ gama_g,
                    const float* __restrict__ fcw_g,
                    const float* __restrict__ fcb_g,
                    float* __restrict__ out) {
  __shared__ alignas(16) unsigned char lds_raw[LDS_BYTES];

  const int bl   = blockIdx.x;
  const int b    = bl / kL;
  const int l    = bl % kL;
  const int tid  = threadIdx.x;
  const int wave = tid >> 5;
  const int lane = tid & 31;
  const int g    = lane >> 4;   // half-wave group
  const int hl   = lane & 15;   // lane within half
  const int nb   = wave * 16;   // this wave's row strip

  __bf16* const wqt   = (__bf16*)(lds_raw);
  __bf16* const wkt   = (__bf16*)(lds_raw + 33792);
  __bf16* const wvt   = (__bf16*)(lds_raw + 2 * 33792);
  __bf16* const stage = (__bf16*)(lds_raw + wave * 2304);
  __bf16* const fcs   = (__bf16*)(lds_raw + 36864);
  __bf16* const Qs    = (__bf16*)(lds_raw + 101376);
  __bf16* const Ks    = (__bf16*)(lds_raw + 101376 + 36864);
  __bf16* const Vt    = (__bf16*)(lds_raw + 101376 + 2 * 36864);

  // ---- stage 0: projection weights for this l, transposed, f32 -> bf16 ----
  {
    const float* wq = Wq_g + (size_t)l * kC * kA;
    const float* wk = Wk_g + (size_t)l * kC * kA;
    const float* wv = Wv_g + (size_t)l * kC * kA;
    for (int idx = tid; idx < kC * kA; idx += kThreads) {
      const int c = idx >> 6, a = idx & 63;
      wqt[a * WT_STRIDE + c] = (__bf16)wq[idx];
      wkt[a * WT_STRIDE + c] = (__bf16)wk[idx];
      wvt[a * WT_STRIDE + c] = (__bf16)wv[idx];
    }
  }
  __syncthreads();

  // ---- phase 1: Q,K,V = X @ W + bias ------------------------------------
  const float* yrow = Y + ((size_t)(b * kN + nb + hl) * kL + l) * kC;
  v8f accQ[4], accK[4], accV[4];
  {
    const v8f z = {};
#pragma unroll
    for (int t = 0; t < 4; ++t) { accQ[t] = z; accK[t] = z; accV[t] = z; }
  }
  for (int kb = 0; kb < kC; kb += 32) {
    if (kb + 32 < kC) __builtin_prefetch(yrow + kb + 32, 0, 3);
    const float4 x0 = *(const float4*)(yrow + kb + g * 8);
    const float4 x1 = *(const float4*)(yrow + kb + g * 8 + 4);
    const float4 x2 = *(const float4*)(yrow + kb + 16 + g * 8);
    const float4 x3 = *(const float4*)(yrow + kb + 16 + g * 8 + 4);
    v16bf a;
    a[0]=(__bf16)x0.x; a[1]=(__bf16)x0.y; a[2]=(__bf16)x0.z; a[3]=(__bf16)x0.w;
    a[4]=(__bf16)x1.x; a[5]=(__bf16)x1.y; a[6]=(__bf16)x1.z; a[7]=(__bf16)x1.w;
    a[8]=(__bf16)x2.x; a[9]=(__bf16)x2.y; a[10]=(__bf16)x2.z; a[11]=(__bf16)x2.w;
    a[12]=(__bf16)x3.x; a[13]=(__bf16)x3.y; a[14]=(__bf16)x3.z; a[15]=(__bf16)x3.w;
    const int boff = kb + g * 16;
#pragma unroll
    for (int t = 0; t < 4; ++t) {
      const int arow = (16 * t + hl) * WT_STRIDE;
      // issue all three fragment loads before consuming any -> partial waits
      const v16bf bq = load_b(wqt + arow, boff);
      const v16bf bk = load_b(wkt + arow, boff);
      const v16bf bv = load_b(wvt + arow, boff);
      accQ[t] = wmma_bf16(a, bq, accQ[t]);
      accK[t] = wmma_bf16(a, bk, accK[t]);
      accV[t] = wmma_bf16(a, bv, accV[t]);
    }
  }
  // bias + write to LDS (Q,K row-major; V transposed)
#pragma unroll
  for (int t = 0; t < 4; ++t) {
    const float qb = Qb_g[16 * t + hl];
    const float kb2 = Kb_g[16 * t + hl];
    const float vb = Vb_g[16 * t + hl];
#pragma unroll
    for (int r = 0; r < 8; ++r) {
      const int row = nb + r + 8 * g;
      Qs[row * QK_STRIDE + 16 * t + hl] = (__bf16)(accQ[t][r] + qb);
      Ks[row * QK_STRIDE + 16 * t + hl] = (__bf16)(accK[t][r] + kb2);
      Vt[(16 * t + hl) * WT_STRIDE + row] = (__bf16)(accV[t][r] + vb);
    }
  }
  __syncthreads();

  // fc weight into LDS (row-major [c][a]); overlays dead W_t region
  for (int idx = tid; idx < kC * kA; idx += kThreads) {
    const int c = idx >> 6, a = idx & 63;
    fcs[c * QK_STRIDE + a] = (__bf16)fcw_g[idx];
  }
  __syncthreads();

  // ---- phase 2: S = Q @ K^T, strip of 16 rows x 256 cols in registers ----
  v8f accS[16];
  {
    const v8f z = {};
#pragma unroll
    for (int mt = 0; mt < 16; ++mt) accS[mt] = z;
  }
  for (int kb = 0; kb < kA; kb += 32) {
    const v16bf aq = load_a(Qs + (nb + hl) * QK_STRIDE, kb, g);
    const int boff = kb + g * 16;
    v16bf bcur = load_b(Ks + hl * QK_STRIDE, boff);  // mt = 0
#pragma unroll
    for (int mt = 0; mt < 16; ++mt) {
      v16bf bnxt = bcur;
      if (mt < 15)
        bnxt = load_b(Ks + ((mt + 1) * 16 + hl) * QK_STRIDE, boff);
      accS[mt] = wmma_bf16(aq, bcur, accS[mt]);
      bcur = bnxt;
    }
  }

  // ---- softmax over 256-wide rows (16 lanes/halfwave x 16 tiles) ---------
#pragma unroll
  for (int r = 0; r < 8; ++r) {
    float m = -3.0e38f;
#pragma unroll
    for (int mt = 0; mt < 16; ++mt) m = fmaxf(m, accS[mt][r]);
    for (int s = 1; s < 16; s <<= 1) m = fmaxf(m, __shfl_xor(m, s, 16));
    float sum = 0.0f;
#pragma unroll
    for (int mt = 0; mt < 16; ++mt) {
      const float e = __expf(accS[mt][r] - m);
      accS[mt][r] = e;
      sum += e;
    }
    for (int s = 1; s < 16; s <<= 1) sum += __shfl_xor(sum, s, 16);
    const float inv = 1.0f / sum;
#pragma unroll
    for (int mt = 0; mt < 16; ++mt) accS[mt][r] *= inv;
  }

  // ---- phase 3: O = P @ V (P transposed C/D->A via per-wave LDS stage) ---
  v8f accO[4];
  {
    const v8f z = {};
#pragma unroll
    for (int t = 0; t < 4; ++t) accO[t] = z;
  }
  for (int j = 0; j < 8; ++j) {  // K chunk = cols 32j..32j+31 of P
#pragma unroll
    for (int tt = 0; tt < 2; ++tt)
#pragma unroll
      for (int r = 0; r < 8; ++r)
        stage[(r + 8 * g) * STG_P_STRIDE + 16 * tt + hl] =
            (__bf16)accS[2 * j + tt][r];
    asm volatile("s_wait_dscnt 0x0" ::: "memory");  // wave-local RAW
    const v16bf ap = load_a(stage + hl * STG_P_STRIDE, 0, g);
    const int boff = 32 * j + g * 16;
    // issue all four V fragment loads, then the four WMMAs
    const v16bf bv0 = load_b(Vt + (0 * 16 + hl) * WT_STRIDE, boff);
    const v16bf bv1 = load_b(Vt + (1 * 16 + hl) * WT_STRIDE, boff);
    const v16bf bv2 = load_b(Vt + (2 * 16 + hl) * WT_STRIDE, boff);
    const v16bf bv3 = load_b(Vt + (3 * 16 + hl) * WT_STRIDE, boff);
    accO[0] = wmma_bf16(ap, bv0, accO[0]);
    accO[1] = wmma_bf16(ap, bv1, accO[1]);
    accO[2] = wmma_bf16(ap, bv2, accO[2]);
    accO[3] = wmma_bf16(ap, bv3, accO[3]);
  }

  // ---- phase 4: F = O @ fc_w^T ------------------------------------------
#pragma unroll
  for (int t = 0; t < 4; ++t)
#pragma unroll
    for (int r = 0; r < 8; ++r)
      stage[(r + 8 * g) * QK_STRIDE + 16 * t + hl] = (__bf16)accO[t][r];
  asm volatile("s_wait_dscnt 0x0" ::: "memory");

  v8f accF[16];
  {
    const v8f z = {};
#pragma unroll
    for (int ct = 0; ct < 16; ++ct) accF[ct] = z;
  }
  for (int kb = 0; kb < kA; kb += 32) {
    const v16bf ao = load_a(stage + hl * QK_STRIDE, kb, g);
    const int boff = kb + g * 16;
    v16bf bcur = load_b(fcs + hl * QK_STRIDE, boff);  // ct = 0
#pragma unroll
    for (int ct = 0; ct < 16; ++ct) {
      v16bf bnxt = bcur;
      if (ct < 15)
        bnxt = load_b(fcs + ((ct + 1) * 16 + hl) * QK_STRIDE, boff);
      accF[ct] = wmma_bf16(ao, bcur, accF[ct]);
      bcur = bnxt;
    }
  }

  // ---- epilogue: out = Y + gama * (F + fc_b) ----------------------------
  const float gm = gama_g[0];
#pragma unroll
  for (int ct = 0; ct < 16; ++ct) {
    const int c = 16 * ct + hl;
    const float fb = fcb_g[c];
#pragma unroll
    for (int r = 0; r < 8; ++r) {
      const int n = nb + r + 8 * g;
      const size_t off = ((size_t)(b * kN + n) * kL + l) * kC + c;
      out[off] = Y[off] + gm * (accF[ct][r] + fb);
    }
  }
}

extern "C" void kernel_launch(void* const* d_in, const int* in_sizes, int n_in,
                              void* d_out, int out_size, void* d_ws, size_t ws_size,
                              hipStream_t stream) {
  const float* Y    = (const float*)d_in[0];
  const float* Wq   = (const float*)d_in[1];
  const float* Wk   = (const float*)d_in[2];
  const float* Wv   = (const float*)d_in[3];
  const float* Qb   = (const float*)d_in[4];
  const float* Kb   = (const float*)d_in[5];
  const float* Vb   = (const float*)d_in[6];
  const float* gama = (const float*)d_in[7];
  const float* fcw  = (const float*)d_in[8];
  const float* fcb  = (const float*)d_in[9];
  float* out = (float*)d_out;

  dim3 grid(kB * kL);
  dim3 block(kThreads);
  va_attn_kernel<<<grid, block, 0, stream>>>(Y, Wq, Wk, Wv, Qb, Kb, Vb, gama,
                                             fcw, fcb, out);
}
